// IsolatedFarFieldLongRange_66881230733960
// MI455X (gfx1250) — compile-verified, matrix operand order checked by reading, not run on hardware
//
#include <hip/hip_runtime.h>
#include <hip/hip_bf16.h>
#include <math.h>

// ---------------------------------------------------------------------------
// CDNA5 (gfx1250) types for WMMA
// ---------------------------------------------------------------------------
typedef __bf16 bf16_t;
typedef __attribute__((ext_vector_type(16))) __bf16 v16bf;
typedef __attribute__((ext_vector_type(8)))  __bf16 v8bf;
typedef __attribute__((ext_vector_type(8)))  float  v8f;

__device__ __forceinline__ float silu_f(float x)    { return x / (1.0f + __expf(-x)); }
__device__ __forceinline__ float sigmoid_f(float x) { return 1.0f / (1.0f + __expf(-x)); }

// A-fragment (16x32 bf16) from an LDS tile, row-major with `stride` elements.
// ISA layout: lanes 0-15 hold row M=lane, K = k0+{0..7, 16..23};
//             lanes 16-31 hold row M=lane-16, K = k0+{8..15, 24..31}.
__device__ __forceinline__ v16bf lds_a_frag(const bf16_t* tile, int stride, int k0, int lane) {
  int r    = lane & 15;
  int koff = (lane < 16) ? 0 : 8;
  const bf16_t* p = tile + r * stride + k0 + koff;     // 16B aligned (k0,koff mult of 8)
  v8bf lo = *(const v8bf*)(p);
  v8bf hi = *(const v8bf*)(p + 16);
  return __builtin_shufflevector(lo, hi, 0,1,2,3,4,5,6,7,8,9,10,11,12,13,14,15);
}

// B-fragment from a fragment-major packed table: frag f, lane l at
// P[(f*32+l)*16 .. +15].  One 32-byte per-lane load, coalesced, zero packing.
__device__ __forceinline__ v16bf packed_b_frag(const bf16_t* __restrict__ P, int frag, int lane) {
  return *(const v16bf*)(P + (((size_t)frag * 32 + lane) << 4));
}

// fragment-major packing index helper: K index for (lane, j) inside a 32-K frag
__device__ __forceinline__ int frag_k(int lane, int j) {
  int koff = (lane < 16) ? 0 : 8;
  return (j < 8) ? (koff + j) : (16 + koff + (j - 8));
}

// ---------------------------------------------------------------------------
// Kernel 0: zero the atomic-accumulator region (must be re-zeroed every call)
// ---------------------------------------------------------------------------
__global__ void k_zero(float* __restrict__ p, long n) {
  long i = (long)blockIdx.x * blockDim.x + threadIdx.x;
  if (i < n) p[i] = 0.f;
}

// ---------------------------------------------------------------------------
// Kernel 1: weight prep.  LN folded into weights:
//   LN(x)W = rs*(x (g.*W)) - m*rs*u + v,  u = sum_k g_k W_k, v = sum_k b_k W_k
// B matrices are swizzled to fragment-major bf16 so each lane's WMMA operand
// is one contiguous 32-byte load:
//   w1pk : (g .* src_w1)            32 frags (8 kt x 4 nb)  [256x64]
//   w2pk : src_w2                    2 frags (2 kt x 1 nb)  [64x16]
//   ehpk : (ln_e_g .* eh_w1) pad 96 12 frags (3 kt x 4 nb)  [96x64]
// ---------------------------------------------------------------------------
__global__ void k_prep(const float* __restrict__ w1,  const float* __restrict__ lng,
                       const float* __restrict__ lnb, const float* __restrict__ w2,
                       const float* __restrict__ ehw1, const float* __restrict__ leg,
                       const float* __restrict__ leb,
                       bf16_t* __restrict__ w1pk, float* __restrict__ u1, float* __restrict__ v1,
                       bf16_t* __restrict__ w2pk, bf16_t* __restrict__ ehpk,
                       float* __restrict__ ue, float* __restrict__ ve) {
  int idx = blockIdx.x * blockDim.x + threadIdx.x;
  int lane = (idx >> 4) & 31;
  int j    = idx & 15;
  if (idx < 32 * 512) {                       // w1 fragments
    int f = idx >> 9, kt = f >> 2, nb = f & 3;
    int n = nb * 16 + (lane & 15);
    int k = kt * 32 + frag_k(lane, j);
    w1pk[idx] = (bf16_t)(lng[k] * w1[k * 64 + n]);
  }
  if (idx < 2 * 512) {                        // w2 fragments
    int kt = idx >> 9;
    int n = lane & 15;
    int k = kt * 32 + frag_k(lane, j);
    w2pk[idx] = (bf16_t)w2[k * 16 + n];
  }
  if (idx < 12 * 512) {                       // energy-head fragments (K pad 67->96)
    int f = idx >> 9, kt = f >> 2, nb = f & 3;
    int n = nb * 16 + (lane & 15);
    int k = kt * 32 + frag_k(lane, j);
    ehpk[idx] = (bf16_t)((k < 67) ? leg[k] * ehw1[k * 64 + n] : 0.f);
  }
  if (idx < 64) {                             // rank-1 LN correction vectors
    float a = 0.f, b = 0.f;
    for (int k = 0; k < 256; ++k) { a += lng[k] * w1[k * 64 + idx]; b += lnb[k] * w1[k * 64 + idx]; }
    u1[idx] = a; v1[idx] = b;
    float c = 0.f, d = 0.f;
    for (int k = 0; k < 67; ++k) { c += leg[k] * ehw1[k * 64 + idx]; d += leb[k] * ehw1[k * 64 + idx]; }
    ue[idx] = c; ve[idx] = d;
  }
}

// ---------------------------------------------------------------------------
// Kernel 2: fused source head.  LN -> [256x64] GEMM -> SiLU -> [64x16] GEMM.
// 128 threads = 4 waves; each wave owns a 16-atom tile.  Features are read
// from HBM exactly once (the 102 MB roofline term), staged raw as bf16 in
// LDS while fp32 LN stats are computed, then consumed by v_wmma_f32_16x16x32_bf16.
// ---------------------------------------------------------------------------
__global__ void __launch_bounds__(128) k_source(
    const float* __restrict__ x, const bf16_t* __restrict__ w1pk,
    const float* __restrict__ u1, const float* __restrict__ v1,
    const float* __restrict__ b1, const bf16_t* __restrict__ w2pk,
    const float* __restrict__ b2, float* __restrict__ src_raw, int N) {
  __shared__ __align__(16) bf16_t s_x[4][16 * 256];   // raw features, bf16
  __shared__ __align__(16) bf16_t s_h[4][16 * 64];    // silu(hidden), bf16
  __shared__ float s_mean[4][16];
  __shared__ float s_rs[4][16];

  const int w    = threadIdx.x >> 5;
  const int lane = threadIdx.x & 31;
  const int r    = lane & 15;
  const int half = lane >> 4;
  const int row0 = blockIdx.x * 64 + w * 16;
  const int row  = min(row0 + r, N - 1);

  // warm the (L2-hot) packed weight table
  __builtin_prefetch((const char*)w1pk + (size_t)threadIdx.x * 256, 0, 0);

  // Phase A: load 128 features per lane (2 lanes per row), bf16-stage + stats
  const float4* p = (const float4*)(x) + (size_t)row * 64 + half * 32;
  bf16_t* dst = &s_x[w][r * 256 + half * 128];
  float sum = 0.f, sq = 0.f;
#pragma unroll
  for (int j = 0; j < 32; ++j) {
    float4 v = p[j];
    sum += v.x + v.y + v.z + v.w;
    sq  += v.x * v.x + v.y * v.y + v.z * v.z + v.w * v.w;
    dst[j * 4 + 0] = (bf16_t)v.x; dst[j * 4 + 1] = (bf16_t)v.y;
    dst[j * 4 + 2] = (bf16_t)v.z; dst[j * 4 + 3] = (bf16_t)v.w;
  }
  sum += __shfl_xor(sum, 16, 32);
  sq  += __shfl_xor(sq, 16, 32);
  float mean = sum * (1.0f / 256.0f);
  float var  = sq * (1.0f / 256.0f) - mean * mean;
  if (half == 0) { s_mean[w][r] = mean; s_rs[w][r] = rsqrtf(var + 1e-5f); }
  __syncthreads();

  // Phase B: GEMM1, 16x64 tile, K=256 in 8 steps of 32
  v8f acc[4] = {};
#pragma unroll
  for (int kt = 0; kt < 8; ++kt) {
    v16bf a = lds_a_frag(&s_x[w][0], 256, kt * 32, lane);
#pragma unroll
    for (int nb = 0; nb < 4; ++nb) {
      v16bf b = packed_b_frag(w1pk, kt * 4 + nb, lane);
      acc[nb] = __builtin_amdgcn_wmma_f32_16x16x32_bf16(false, a, false, b,
                                                        (short)0, acc[nb], false, false);
    }
  }

  // Phase C: fp32 LN correction + bias + SiLU, restage bf16
#pragma unroll
  for (int nb = 0; nb < 4; ++nb) {
#pragma unroll
    for (int i = 0; i < 8; ++i) {
      int m = i + half * 8;                 // C layout: VGPR i -> row i / i+8
      int n = nb * 16 + r;
      float mm = s_mean[w][m], rr = s_rs[w][m];
      float y = rr * acc[nb][i] - mm * rr * u1[n] + v1[n] + b1[n];
      s_h[w][m * 64 + n] = (bf16_t)silu_f(y);
    }
  }
  __syncthreads();

  // Phase D: GEMM2 (64 -> 16), K=64 in 2 steps
  v8f acc2 = {};
#pragma unroll
  for (int kt = 0; kt < 2; ++kt) {
    v16bf a = lds_a_frag(&s_h[w][0], 64, kt * 32, lane);
    v16bf b = packed_b_frag(w2pk, kt, lane);
    acc2 = __builtin_amdgcn_wmma_f32_16x16x32_bf16(false, a, false, b,
                                                   (short)0, acc2, false, false);
  }
#pragma unroll
  for (int i = 0; i < 8; ++i) {
    int m = i + half * 8;
    int ro = row0 + m;
    if (ro < N) src_raw[(size_t)ro * 16 + r] = acc2[i] + b2[r];
  }
}

// ---------------------------------------------------------------------------
// Kernel 3: per-graph sums (counts, source sum, pos sum) via fp32 atomics
// ---------------------------------------------------------------------------
__global__ void k_gsum(const float* __restrict__ src_raw, const float* __restrict__ pos,
                       const int* __restrict__ batch, float* __restrict__ g_sum,
                       float* __restrict__ g_cnt, float* __restrict__ g_pos, int N) {
  int idx = blockIdx.x * blockDim.x + threadIdx.x;
  if (idx >= N * 16) return;
  int a = idx >> 4, c = idx & 15;
  int g = batch[a];
  atomicAdd(&g_sum[g * 16 + c], src_raw[idx]);
  if (c == 0) atomicAdd(&g_cnt[g], 1.f);
  if (c < 3)  atomicAdd(&g_pos[g * 3 + c], pos[(size_t)a * 3 + c]);
}

// ---------------------------------------------------------------------------
// Kernel 4: center source, accumulate per-graph centered sum / sumsq / radial
// ---------------------------------------------------------------------------
__global__ void k_center(const float* __restrict__ src_raw, const float* __restrict__ pos,
                         const int* __restrict__ batch, const float* __restrict__ g_sum,
                         const float* __restrict__ g_cnt, const float* __restrict__ g_pos,
                         float* __restrict__ source, float* __restrict__ g_sumc,
                         float* __restrict__ g_sq, float* __restrict__ radial,
                         float* __restrict__ g_rad, int N) {
  int idx = blockIdx.x * blockDim.x + threadIdx.x;
  if (idx >= N * 16) return;
  int a = idx >> 4, c = idx & 15;
  int g = batch[a];
  float cn = fmaxf(g_cnt[g], 1.f);
  float s = src_raw[idx] - g_sum[g * 16 + c] / cn;
  source[idx] = s;
  atomicAdd(&g_sumc[g * 16 + c], s);
  atomicAdd(&g_sq[g * 16 + c], s * s);
  if (c == 0) {
    float dx = pos[(size_t)a * 3 + 0] - g_pos[g * 3 + 0] / cn;
    float dy = pos[(size_t)a * 3 + 1] - g_pos[g * 3 + 1] / cn;
    float dz = pos[(size_t)a * 3 + 2] - g_pos[g * 3 + 2] / cn;
    float rr = sqrtf(dx * dx + dy * dy + dz * dz);
    radial[a] = rr;
    atomicAdd(&g_rad[g], rr);
  }
}

// ---------------------------------------------------------------------------
// Kernel 5: edge scatter.  source is 6.4 MB (L2-resident on 192 MB L2);
// 66M fp32 L2 atomics — the L2-bound cost center of the model.
// ---------------------------------------------------------------------------
__global__ void k_edge(const float* __restrict__ source, const int* __restrict__ esrc,
                       const int* __restrict__ edst, float* __restrict__ nbr_sum,
                       float* __restrict__ nbr_sq, float* __restrict__ deg, int E) {
  long idx = (long)blockIdx.x * blockDim.x + threadIdx.x;
  if (idx >= (long)E * 16) return;
  int e = (int)(idx >> 4), c = (int)(idx & 15);
  int s = esrc[e], d = edst[e];
  float v = source[(size_t)s * 16 + c];
  atomicAdd(&nbr_sum[(size_t)d * 16 + c], v);
  atomicAdd(&nbr_sq[(size_t)d * 16 + c], v * v);
  if (c == 0) atomicAdd(&deg[d], 1.f);
}

// ---------------------------------------------------------------------------
// Kernel 6: fused final head.  far-field stats -> shell MLP (VALU) ->
// fused 67-vector staged bf16 in LDS (LN folded into ehpk) ->
// WMMA [16x96]x[96x64] -> SiLU -> dot(eh_w2) with shfl_xor wave reduction.
// ---------------------------------------------------------------------------
__global__ void __launch_bounds__(128) k_final(
    const float* __restrict__ source, const float* __restrict__ nbr_sum,
    const float* __restrict__ nbr_sq, const float* __restrict__ deg,
    const float* __restrict__ radial, const int* __restrict__ batch,
    const float* __restrict__ g_cnt, const float* __restrict__ g_sumc,
    const float* __restrict__ g_sq, const float* __restrict__ g_rad,
    const float* __restrict__ sh_w1, const float* __restrict__ sh_b1,
    const float* __restrict__ sh_w2, const float* __restrict__ sh_b2,
    const bf16_t* __restrict__ ehpk, const float* __restrict__ ue,
    const float* __restrict__ ve, const float* __restrict__ eh_b1,
    const float* __restrict__ eh_w2, const float* __restrict__ eh_b2,
    const float* __restrict__ escale, float* __restrict__ out, int N) {
  __shared__ __align__(16) bf16_t s_fb[4][16 * 96];   // fused vector, bf16, K-padded
  __shared__ float s_hid[4][16 * 64];                 // shell MLP hidden
  __shared__ float s_mean[4][16];
  __shared__ float s_rs[4][16];

  const int w    = threadIdx.x >> 5;
  const int lane = threadIdx.x & 31;
  const int r    = lane & 15;
  const int half = lane >> 4;
  const int row0 = blockIdx.x * 64 + w * 16;
  const int a    = min(row0 + r, N - 1);

  const int   g    = batch[a];
  const float cntg = g_cnt[g];
  const float cn   = fmaxf(cntg, 1.f);
  const float lc   = 1.f + deg[a];
  const float fc   = fmaxf(cntg - lc, 0.f);
  const float fd   = fmaxf(fc, 1.f);
  const float st0  = log1pf(fc);
  const float st1  = fc / fmaxf(cntg, 1.f);
  const float rsc  = fmaxf(g_rad[g] / cn, 1e-6f);
  const float st2  = radial[a] / rsc;

  // shell MLP hidden (3 -> 64); two half-lanes per atom split the 64 units
#pragma unroll
  for (int jj = 0; jj < 32; ++jj) {
    int j = half * 32 + jj;
    float h = st0 * sh_w1[j] + st1 * sh_w1[64 + j] + st2 * sh_w1[128 + j] + sh_b1[j];
    s_hid[w][r * 64 + j] = silu_f(h);
  }
  __syncthreads();

  float sv[8], fmv[8], fgv[8];
  float sum = 0.f, sq = 0.f;
#pragma unroll
  for (int c = 0; c < 8; ++c) {
    int cc = half * 8 + c;
    float s  = source[(size_t)a * 16 + cc];
    float ls = s + nbr_sum[(size_t)a * 16 + cc];
    float lq = s * s + nbr_sq[(size_t)a * 16 + cc];
    float fm = (g_sumc[g * 16 + cc] - ls) / fd;
    float fv = fmaxf((g_sq[g * 16 + cc] - lq) / fd - fm * fm, 0.f);
    float fs = sqrtf(fv + 1e-8f);
    float gate = sh_b2[cc];
#pragma unroll 4
    for (int j = 0; j < 64; ++j) gate += s_hid[w][r * 64 + j] * sh_w2[j * 16 + cc];
    gate = sigmoid_f(gate);
    sv[c] = s; fmv[c] = fm; fgv[c] = fs * gate;
  }

  // stage fused row bf16 + fp32 LN stats over the 67 real features
  bf16_t* frow = &s_fb[w][r * 96];
#pragma unroll
  for (int c = 0; c < 8; ++c) {
    int cc = half * 8 + c;
    float a0 = sv[c], a1 = fmv[c], a2 = sv[c] * fmv[c], a3 = fgv[c];
    frow[cc] = (bf16_t)a0; frow[16 + cc] = (bf16_t)a1;
    frow[32 + cc] = (bf16_t)a2; frow[48 + cc] = (bf16_t)a3;
    sum += a0 + a1 + a2 + a3;
    sq  += a0 * a0 + a1 * a1 + a2 * a2 + a3 * a3;
  }
  if (half == 0) {
    frow[64] = (bf16_t)st0; frow[65] = (bf16_t)st1; frow[66] = (bf16_t)st2;
    sum += st0 + st1 + st2;
    sq  += st0 * st0 + st1 * st1 + st2 * st2;
#pragma unroll
    for (int j = 67; j < 80; ++j) frow[j] = (bf16_t)0.f;   // K padding
  } else {
#pragma unroll
    for (int j = 80; j < 96; ++j) frow[j] = (bf16_t)0.f;
  }
  sum += __shfl_xor(sum, 16, 32);
  sq  += __shfl_xor(sq, 16, 32);
  float mean = sum / 67.f;
  float var  = sq / 67.f - mean * mean;
  if (half == 0) { s_mean[w][r] = mean; s_rs[w][r] = rsqrtf(var + 1e-5f); }
  __syncthreads();

  // energy-head GEMM: [16 x 96(bf16)] x [96 x 64], K padded (zero weight rows)
  v8f acc[4] = {};
#pragma unroll
  for (int kt = 0; kt < 3; ++kt) {
    v16bf af = lds_a_frag(&s_fb[w][0], 96, kt * 32, lane);
#pragma unroll
    for (int nb = 0; nb < 4; ++nb) {
      v16bf bf = packed_b_frag(ehpk, kt * 4 + nb, lane);
      acc[nb] = __builtin_amdgcn_wmma_f32_16x16x32_bf16(false, af, false, bf,
                                                        (short)0, acc[nb], false, false);
    }
  }

  // LN correction + bias + SiLU + dot with eh_w2, wave reduce, write [N,1]
  const float es  = escale[0];
  const float eb2 = eh_b2[0];
#pragma unroll
  for (int i = 0; i < 8; ++i) {
    int m = i + half * 8;
    float mm = s_mean[w][m], rr = s_rs[w][m];
    float part = 0.f;
#pragma unroll
    for (int nb = 0; nb < 4; ++nb) {
      int n = nb * 16 + r;
      float y = rr * acc[nb][i] - mm * rr * ue[n] + ve[n] + eh_b1[n];
      part += silu_f(y) * eh_w2[n];
    }
    part += __shfl_xor(part, 1, 32);
    part += __shfl_xor(part, 2, 32);
    part += __shfl_xor(part, 4, 32);
    part += __shfl_xor(part, 8, 32);
    if (r == 0) {
      int row = row0 + m;
      if (row < N) out[row] = es * (part + eb2);
    }
  }
}

// ---------------------------------------------------------------------------
// Host-side orchestration
// ---------------------------------------------------------------------------
extern "C" void kernel_launch(void* const* d_in, const int* in_sizes, int n_in,
                              void* d_out, int out_size, void* d_ws, size_t ws_size,
                              hipStream_t stream) {
  const float* x     = (const float*)d_in[0];
  const float* pos   = (const float*)d_in[1];
  const float* lng   = (const float*)d_in[3];
  const float* lnb   = (const float*)d_in[4];
  const float* w1    = (const float*)d_in[5];
  const float* b1    = (const float*)d_in[6];
  const float* w2    = (const float*)d_in[7];
  const float* b2    = (const float*)d_in[8];
  const float* shw1  = (const float*)d_in[9];
  const float* shb1  = (const float*)d_in[10];
  const float* shw2  = (const float*)d_in[11];
  const float* shb2  = (const float*)d_in[12];
  const float* leg   = (const float*)d_in[13];
  const float* leb   = (const float*)d_in[14];
  const float* ehw1  = (const float*)d_in[15];
  const float* ehb1  = (const float*)d_in[16];
  const float* ehw2  = (const float*)d_in[17];
  const float* ehb2  = (const float*)d_in[18];
  const float* esc   = (const float*)d_in[19];
  const int*   batch = (const int*)d_in[20];
  const int*   esrc  = (const int*)d_in[21];
  const int*   edst  = (const int*)d_in[22];
  const int N = in_sizes[20];
  const int E = in_sizes[21];
  const int G = in_sizes[2] / 9;

  char* ws = (char*)d_ws;
  size_t off = 0;
  auto alloc = [&](size_t bytes) -> void* {
    void* p = ws + off;
    off = (off + bytes + 255) & ~(size_t)255;
    return p;
  };
  // non-zeroed region (fully overwritten each call)
  bf16_t* w1pk    = (bf16_t*)alloc(32 * 512 * sizeof(bf16_t));   // 32 frags
  bf16_t* w2pk    = (bf16_t*)alloc(2 * 512 * sizeof(bf16_t));    // 2 frags
  bf16_t* ehpk    = (bf16_t*)alloc(12 * 512 * sizeof(bf16_t));   // 12 frags
  float*  u1      = (float*)alloc(64 * sizeof(float));
  float*  v1      = (float*)alloc(64 * sizeof(float));
  float*  ue      = (float*)alloc(64 * sizeof(float));
  float*  ve      = (float*)alloc(64 * sizeof(float));
  float*  src_raw = (float*)alloc((size_t)N * 16 * sizeof(float));
  float*  source  = (float*)alloc((size_t)N * 16 * sizeof(float));
  float*  radial  = (float*)alloc((size_t)N * sizeof(float));
  // zeroed accumulator region (contiguous)
  size_t zoff = off;
  float* nbr_sum = (float*)alloc((size_t)N * 16 * sizeof(float));
  float* nbr_sq  = (float*)alloc((size_t)N * 16 * sizeof(float));
  float* deg     = (float*)alloc((size_t)N * sizeof(float));
  float* g_cnt   = (float*)alloc((size_t)G * sizeof(float));
  float* g_sum   = (float*)alloc((size_t)G * 16 * sizeof(float));
  float* g_pos   = (float*)alloc((size_t)G * 3 * sizeof(float));
  float* g_sumc  = (float*)alloc((size_t)G * 16 * sizeof(float));
  float* g_sq    = (float*)alloc((size_t)G * 16 * sizeof(float));
  float* g_rad   = (float*)alloc((size_t)G * sizeof(float));
  long zwords = (long)((off - zoff) / sizeof(float));

  k_zero<<<(int)((zwords + 255) / 256), 256, 0, stream>>>((float*)(ws + zoff), zwords);

  k_prep<<<64, 256, 0, stream>>>(w1, lng, lnb, w2, ehw1, leg, leb,
                                 w1pk, u1, v1, w2pk, ehpk, ue, ve);

  k_source<<<(N + 63) / 64, 128, 0, stream>>>(x, w1pk, u1, v1, b1, w2pk, b2, src_raw, N);

  k_gsum<<<(N * 16 + 255) / 256, 256, 0, stream>>>(src_raw, pos, batch, g_sum, g_cnt, g_pos, N);

  k_center<<<(N * 16 + 255) / 256, 256, 0, stream>>>(src_raw, pos, batch, g_sum, g_cnt, g_pos,
                                                     source, g_sumc, g_sq, radial, g_rad, N);

  k_edge<<<(int)(((long)E * 16 + 255) / 256), 256, 0, stream>>>(source, esrc, edst,
                                                                nbr_sum, nbr_sq, deg, E);

  k_final<<<(N + 63) / 64, 128, 0, stream>>>(source, nbr_sum, nbr_sq, deg, radial, batch,
                                             g_cnt, g_sumc, g_sq, g_rad,
                                             shw1, shb1, shw2, shb2,
                                             ehpk, ue, ve, ehb1, ehw2, ehb2, esc,
                                             (float*)d_out, N);
}
